// EIGLayerComplex_67997922230878
// MI455X (gfx1250) — matrix operand order repaired and checked by reading, not compiled, not run on hardware
//
#include <hip/hip_runtime.h>
#include <hip/hip_bf16.h>
#include <stdint.h>

// ---------------------------------------------------------------------------
// EIG / PNA-style GNN layer for gfx1250 (MI455X), wave32 + WMMA f16->f32.
//   K1 init accumulators, K2 transpose+f16-convert weights,
//   K3 fused edge pretrans GEMM (WMMA) + 5-way atomic scatter,
//   K4 finalize aggregators -> f16 [h|agg] matrix,
//   K5 post GEMM (WMMA) + graphnorm + BN column sums,
//   K6 batchnorm + relu + residual.
// ---------------------------------------------------------------------------

typedef _Float16 f16;
typedef __attribute__((ext_vector_type(16))) _Float16     v16h;
typedef __attribute__((ext_vector_type(8)))  float        v8f;
typedef __attribute__((ext_vector_type(4)))  unsigned int uu4;

union H16 { v16h h; uu4 u[2]; };

#define N_NODES 50000
#define N_EDGES 800000
#define DIM     96
#define EDIM    8
#define KPRE    224   // 200 (96+96+8) zero-padded to 7*32
#define KPRE_T  7
#define KPOST   576   // 18*32
#define KPOST_T 18
#define NT      6     // 96 / 16 column tiles
#define NC      (N_NODES * DIM)

// order-preserving float <-> uint encoding for integer atomicMax/Min
__device__ __forceinline__ unsigned int fenc(float f) {
  unsigned int u = __float_as_uint(f);
  return (u & 0x80000000u) ? ~u : (u | 0x80000000u);
}
__device__ __forceinline__ float fdec(unsigned int k) {
  unsigned int u = (k & 0x80000000u) ? (k & 0x7FFFFFFFu) : ~k;
  return __uint_as_float(u);
}

// ---------------- K1: init accumulators ----------------
__global__ void eig_init_kernel(float* __restrict__ s1, float* __restrict__ s2,
                                float* __restrict__ se,
                                unsigned int* __restrict__ mxu,
                                unsigned int* __restrict__ mnu,
                                float* __restrict__ deg, float* __restrict__ eden,
                                float* __restrict__ colsum, float* __restrict__ colsq) {
  int idx = blockIdx.x * blockDim.x + threadIdx.x;
  if (idx < NC) {
    s1[idx] = 0.0f; s2[idx] = 0.0f; se[idx] = 0.0f;
    mxu[idx] = 0u;           // enc() of any finite value is > 0
    mnu[idx] = 0xFFFFFFFFu;
  }
  if (idx < N_NODES) { deg[idx] = 0.0f; eden[idx] = 0.0f; }
  if (idx < DIM) { colsum[idx] = 0.0f; colsq[idx] = 0.0f; }
}

// ---------------- K2: weights -> transposed f16 (K contiguous per column) ----
__global__ void eig_prep_kernel(const float* __restrict__ W_pre,
                                const float* __restrict__ W_post,
                                f16* __restrict__ WpreT, f16* __restrict__ WpostT) {
  int idx = blockIdx.x * blockDim.x + threadIdx.x;
  if (idx >= DIM * KPOST) return;
  int c = idx / KPOST, k = idx % KPOST;
  WpostT[(size_t)c * KPOST + k] = (f16)W_post[(size_t)k * DIM + c];
  if (k < KPRE)
    WpreT[(size_t)c * KPRE + k] = (k < 200) ? (f16)W_pre[(size_t)k * DIM + c]
                                            : (f16)0.0f;
}

// ---------------- K3: fused edge pretrans GEMM + scatter --------------------
// 256 threads = 8 waves, each wave owns 16 edges (E = 50000 waves exactly).
__global__ __launch_bounds__(256) void eig_edge_kernel(
    const float* __restrict__ h, const float* __restrict__ e,
    const float* __restrict__ eig, const int* __restrict__ src,
    const int* __restrict__ dst, const f16* __restrict__ WpreT,
    const float* __restrict__ b_pre,
    float* __restrict__ s1, float* __restrict__ s2, float* __restrict__ se,
    unsigned int* __restrict__ mxu, unsigned int* __restrict__ mnu,
    float* __restrict__ deg, float* __restrict__ eden) {
  __shared__ f16   sA[8 * 16 * KPRE];   // 57344 B: 8 waves x (16 edges x 224) f16
  __shared__ int   sDst[8][16];
  __shared__ float sW[8][16];

  const int tid   = threadIdx.x;
  const int wave  = tid >> 5;
  const int lane  = tid & 31;
  const int ebase = (blockIdx.x * 8 + wave) * 16;

  // ---- stage A tile: [h_src(96) | h_dst(96) | e(8) | pad(24)] as f16 ----
  {
    const int i    = lane >> 1;     // edge within tile: 2 lanes per edge
    const int half = lane & 1;
    const int eidx = ebase + i;
    const int s = src[eidx];
    const int d = dst[eidx];
    f16* row = &sA[(wave * 16 + i) * KPRE];
    const float* hs = h + (size_t)s * DIM + half * 48;
    const float* hd = h + (size_t)d * DIM + half * 48;
#pragma unroll
    for (int c = 0; c < 12; ++c) {
      float4 v = *(const float4*)(hs + c * 4);
      f16* p = row + half * 48 + c * 4;
      p[0] = (f16)v.x; p[1] = (f16)v.y; p[2] = (f16)v.z; p[3] = (f16)v.w;
    }
#pragma unroll
    for (int c = 0; c < 12; ++c) {
      float4 v = *(const float4*)(hd + c * 4);
      f16* p = row + DIM + half * 48 + c * 4;
      p[0] = (f16)v.x; p[1] = (f16)v.y; p[2] = (f16)v.z; p[3] = (f16)v.w;
    }
    {
      float4 v = *(const float4*)(e + (size_t)eidx * EDIM + half * 4);
      f16* p = row + 2 * DIM + half * 4;
      p[0] = (f16)v.x; p[1] = (f16)v.y; p[2] = (f16)v.z; p[3] = (f16)v.w;
    }
    f16* q = row + 200 + half * 12;    // zero pad 200..223
#pragma unroll
    for (int c = 0; c < 12; ++c) q[c] = (f16)0.0f;

    if (lane < 16) {                    // per-edge metadata, deg / eig_den
      const int ei = ebase + lane;
      const int d2 = dst[ei];
      sDst[wave][lane] = d2;
      float wv = fabsf(eig[src[ei]] - eig[d2]);
      sW[wave][lane] = wv;
      unsafeAtomicAdd(&deg[d2], 1.0f);
      unsafeAtomicAdd(&eden[d2], wv);
    }
  }
  __syncthreads();

  // ---- WMMA: m[16x96] = A[16x224] * WpreT^T ----
  const int lanelo = lane & 15;
  const int grp    = lane >> 4;
  v8f acc[NT];
#pragma unroll
  for (int t = 0; t < NT; ++t)
#pragma unroll
    for (int i = 0; i < 8; ++i) acc[t][i] = 0.0f;

  const f16* arow = &sA[(wave * 16 + lanelo) * KPRE];
#pragma unroll
  for (int kt = 0; kt < KPRE_T; ++kt) {
    H16 a;  // A 16x32 f16 fragment: lane (grp,lanelo) -> two 8-half runs
    a.u[0] = *(const uu4*)(arow + kt * 32 + grp * 8);
    a.u[1] = *(const uu4*)(arow + kt * 32 + 16 + grp * 8);
#pragma unroll
    for (int t = 0; t < NT; ++t) {
      const f16* bcol = WpreT + (size_t)(t * 16 + lanelo) * KPRE + kt * 32 + grp * 8;
      H16 b;  // B 32x16 fragment: two 8-half K runs per lane (same as A)
      b.u[0] = *(const uu4*)(bcol);
      b.u[1] = *(const uu4*)(bcol + 16);
      acc[t] = __builtin_amdgcn_wmma_f32_16x16x32_f16(
          false, a.h, false, b.h, (short)0, acc[t], false, false);
    }
  }

  // ---- scatter: sum / sum^2 / eig-weighted / max / min ----
#pragma unroll
  for (int t = 0; t < NT; ++t) {
    const int col = t * 16 + lanelo;
    const float vb = b_pre[col];
#pragma unroll
    for (int i = 0; i < 8; ++i) {
      const int m    = i + grp * 8;        // C layout: VGPR i -> row i + 8*grp
      const int node = sDst[wave][m];
      const float v  = acc[t][i] + vb;
      const size_t off = (size_t)node * DIM + col;
      unsafeAtomicAdd(&s1[off], v);
      unsafeAtomicAdd(&s2[off], v * v);
      unsafeAtomicAdd(&se[off], v * sW[wave][m]);
      atomicMax(&mxu[off], fenc(v));
      atomicMin(&mnu[off], fenc(v));
    }
  }
}

// ---------------- K4: finalize aggregators -> f16 [h | mean|max|min|std|eig] -
__global__ void eig_node_kernel(const float* __restrict__ h,
                                const float* __restrict__ s1,
                                const float* __restrict__ s2,
                                const float* __restrict__ se,
                                const unsigned int* __restrict__ mxu,
                                const unsigned int* __restrict__ mnu,
                                const float* __restrict__ deg,
                                const float* __restrict__ eden,
                                f16* __restrict__ Apost) {
  int idx = blockIdx.x * blockDim.x + threadIdx.x;
  if (idx >= NC) return;
  int n = idx / DIM, c = idx % DIM;
  float dg = deg[n];
  float dc = fmaxf(dg, 1.0f);
  float mean = s1[idx] / dc;
  float msq  = s2[idx] / dc;
  float st   = sqrtf(fmaxf(msq - mean * mean, 0.0f) + 1e-5f);
  float mx   = (dg > 0.5f) ? fdec(mxu[idx]) : 0.0f;   // isfinite(mx) ? mx : 0
  float mn   = (dg > 0.5f) ? fdec(mnu[idx]) : 0.0f;
  float ea   = se[idx] / (eden[n] + 1e-8f);
  f16* row = Apost + (size_t)n * KPOST;
  row[c]       = (f16)h[idx];
  row[96 + c]  = (f16)mean;
  row[192 + c] = (f16)mx;
  row[288 + c] = (f16)mn;
  row[384 + c] = (f16)st;
  row[480 + c] = (f16)ea;
}

// ---------------- K5: post GEMM (WMMA) + bias + graphnorm + BN sums ---------
__global__ __launch_bounds__(256) void eig_post_kernel(
    const f16* __restrict__ Apost, const f16* __restrict__ WpostT,
    const float* __restrict__ b_post, const float* __restrict__ snorm,
    float* __restrict__ outpre, float* __restrict__ colsum,
    float* __restrict__ colsq) {
  const int wid  = (blockIdx.x * blockDim.x + threadIdx.x) >> 5;
  const int lane = threadIdx.x & 31;
  if (wid >= N_NODES / 16) return;     // 3125 tiles exactly, wave-uniform exit
  const int lanelo = lane & 15;
  const int grp    = lane >> 4;
  const int n0     = wid * 16;

  v8f acc[NT];
#pragma unroll
  for (int t = 0; t < NT; ++t)
#pragma unroll
    for (int i = 0; i < 8; ++i) acc[t][i] = 0.0f;

  const f16* arow = Apost + (size_t)(n0 + lanelo) * KPOST;
#pragma unroll
  for (int kt = 0; kt < KPOST_T; ++kt) {
    H16 a;
    a.u[0] = *(const uu4*)(arow + kt * 32 + grp * 8);
    a.u[1] = *(const uu4*)(arow + kt * 32 + 16 + grp * 8);
#pragma unroll
    for (int t = 0; t < NT; ++t) {
      const f16* bcol = WpostT + (size_t)(t * 16 + lanelo) * KPOST + kt * 32 + grp * 8;
      H16 b;
      b.u[0] = *(const uu4*)(bcol);
      b.u[1] = *(const uu4*)(bcol + 16);
      acc[t] = __builtin_amdgcn_wmma_f32_16x16x32_f16(
          false, a.h, false, b.h, (short)0, acc[t], false, false);
    }
  }

#pragma unroll
  for (int t = 0; t < NT; ++t) {
    const int col = t * 16 + lanelo;
    const float vb = b_post[col];
#pragma unroll
    for (int i = 0; i < 8; ++i) {
      const int row = n0 + i + grp * 8;
      float v = (acc[t][i] + vb) * snorm[row];
      outpre[(size_t)row * DIM + col] = v;
      unsafeAtomicAdd(&colsum[col], v);
      unsafeAtomicAdd(&colsq[col], v * v);
    }
  }
}

// ---------------- K6: batchnorm + relu + residual ---------------------------
__global__ void eig_bn_kernel(const float* __restrict__ h,
                              const float* __restrict__ outpre,
                              const float* __restrict__ colsum,
                              const float* __restrict__ colsq,
                              const float* __restrict__ gamma,
                              const float* __restrict__ beta,
                              float* __restrict__ out) {
  int idx = blockIdx.x * blockDim.x + threadIdx.x;
  if (idx >= NC) return;
  int c = idx % DIM;
  const float invN = 1.0f / (float)N_NODES;
  float mu  = colsum[c] * invN;
  float var = colsq[c] * invN - mu * mu;
  float y = (outpre[idx] - mu) * rsqrtf(var + 1e-5f) * gamma[c] + beta[c];
  y = fmaxf(y, 0.0f);
  out[idx] = h[idx] + y;
}

// ---------------------------------------------------------------------------
extern "C" void kernel_launch(void* const* d_in, const int* in_sizes, int n_in,
                              void* d_out, int out_size, void* d_ws, size_t ws_size,
                              hipStream_t stream) {
  const float* h      = (const float*)d_in[0];
  const float* e      = (const float*)d_in[1];
  const float* eig    = (const float*)d_in[2];
  const float* snorm  = (const float*)d_in[3];
  const int*   src    = (const int*)d_in[4];
  const int*   dst    = (const int*)d_in[5];
  const float* W_pre  = (const float*)d_in[6];
  const float* b_pre  = (const float*)d_in[7];
  const float* W_post = (const float*)d_in[8];
  const float* b_post = (const float*)d_in[9];
  const float* gamma  = (const float*)d_in[10];
  const float* beta   = (const float*)d_in[11];
  float* out = (float*)d_out;

  // carve workspace
  char* p = (char*)d_ws;
  auto carve = [&](size_t bytes) {
    char* r = p;
    p += (bytes + 255) & ~(size_t)255;
    return r;
  };
  float*        s1     = (float*)carve(sizeof(float) * NC);
  float*        s2     = (float*)carve(sizeof(float) * NC);
  float*        se     = (float*)carve(sizeof(float) * NC);
  unsigned int* mxu    = (unsigned int*)carve(sizeof(unsigned int) * NC);
  unsigned int* mnu    = (unsigned int*)carve(sizeof(unsigned int) * NC);
  float*        deg    = (float*)carve(sizeof(float) * N_NODES);
  float*        eden   = (float*)carve(sizeof(float) * N_NODES);
  float*        colsum = (float*)carve(sizeof(float) * DIM);
  float*        colsq  = (float*)carve(sizeof(float) * DIM);
  f16*          WpreT  = (f16*)carve(sizeof(f16) * DIM * KPRE);
  f16*          WpostT = (f16*)carve(sizeof(f16) * DIM * KPOST);
  f16*          Apost  = (f16*)carve(sizeof(f16) * (size_t)N_NODES * KPOST);
  float*        outpre = (float*)carve(sizeof(float) * NC);

  const int elemBlocks = (NC + 255) / 256;                 // 18750
  const int prepBlocks = (DIM * KPOST + 255) / 256;        // 216
  const int edgeBlocks = N_EDGES / (8 * 16);               // 6250 exact
  const int postBlocks = (N_NODES / 16 + 7) / 8;           // 391

  eig_init_kernel<<<elemBlocks, 256, 0, stream>>>(s1, s2, se, mxu, mnu, deg, eden,
                                                  colsum, colsq);
  eig_prep_kernel<<<prepBlocks, 256, 0, stream>>>(W_pre, W_post, WpreT, WpostT);
  eig_edge_kernel<<<edgeBlocks, 256, 0, stream>>>(h, e, eig, src, dst, WpreT, b_pre,
                                                  s1, s2, se, mxu, mnu, deg, eden);
  eig_node_kernel<<<elemBlocks, 256, 0, stream>>>(h, s1, s2, se, mxu, mnu, deg, eden,
                                                  Apost);
  eig_post_kernel<<<postBlocks, 256, 0, stream>>>(Apost, WpostT, b_post, snorm,
                                                  outpre, colsum, colsq);
  eig_bn_kernel<<<elemBlocks, 256, 0, stream>>>(h, outpre, colsum, colsq, gamma,
                                                beta, out);
}
